// DynamicDiTBlock_51616916964120
// MI455X (gfx1250) — compile-verified, matrix operand order checked by reading, not run on hardware
//
#include <hip/hip_runtime.h>
#include <hip/hip_bf16.h>

// ---------------------------------------------------------------------------
// DynamicDiT block for MI455X (gfx1250), wave32 + WMMA bf16.
// All matrix math goes through v_wmma_f32_16x16x32_bf16 (f32 accumulate).
// ---------------------------------------------------------------------------

#define B_      4
#define S_      2048
#define D_      1024
#define HEADS_  16
#define HD_     64
#define DFF_    4096
#define TDIM_   256
#define KEEP_   1433                  // int(2048 * 0.7)
#define MTOT_   (B_ * KEEP_)         // 5732 selected rows across batches
#define QT_     ((KEEP_ + 15) / 16)  // 90 q-tiles per batch
#define NCHUNK_ ((KEEP_ + 31) / 32)  // 45 kv chunks of 32

typedef __bf16 v16bf __attribute__((ext_vector_type(16)));
typedef float  v8f   __attribute__((ext_vector_type(8)));

union Frag { v16bf v; uint4 u[2]; };

static __device__ inline v8f wmma_bf16(v16bf a, v16bf b, v8f c) {
  // D = A(16x32 bf16) x B(32x16 bf16) + C(16x16 f32)
  return __builtin_amdgcn_wmma_f32_16x16x32_bf16(
      false, a, false, b, (short)0, c, false, false);
}

// ---------------------------------------------------------------------------
// fp32 -> bf16 conversion (weights / activations staging)
// ---------------------------------------------------------------------------
__global__ void k_cvt(const float* __restrict__ s, __bf16* __restrict__ d, int n) {
  int i = blockIdx.x * 256 + threadIdx.x;
  if (i < n) d[i] = (__bf16)s[i];
}

// ---------------------------------------------------------------------------
// Token importance: silu(x @ w1^T + b1) @ w2^T + b2     (one block per token)
// ---------------------------------------------------------------------------
__global__ __launch_bounds__(128) void k_importance(
    const float* __restrict__ x, const float* __restrict__ w1,
    const float* __restrict__ b1, const float* __restrict__ w2,
    const float* __restrict__ b2, float* __restrict__ imp) {
  __shared__ float red[128];
  const int tok = blockIdx.x;                 // 0 .. B*S-1
  const int t = threadIdx.x;
  const int od = t & 31, part = t >> 5;       // 32 hidden dims x 4 partials
  const float* xr = x + (size_t)tok * D_;
  const float* wr = w1 + (size_t)od * D_;
  float s = 0.f;
  const int d0 = part * 256;
  for (int d = d0; d < d0 + 256; ++d) s += xr[d] * wr[d];
  red[t] = s;
  __syncthreads();
  if (t < 32) {
    float tot = red[t] + red[t + 32] + red[t + 64] + red[t + 96] + b1[od];
    float si = tot / (1.f + __expf(-tot));    // silu
    float val = si * w2[od];
    for (int off = 16; off > 0; off >>= 1) val += __shfl_xor(val, off, 32);
    if (t == 0) imp[tok] = val + b2[0];
  }
}

// ---------------------------------------------------------------------------
// Deterministic top-k selection matching jax.lax.top_k + sort(idx):
// rank(i) = #{ imp[j] > imp[i] } + #{ j < i : imp[j] == imp[i] }; keep rank<K,
// compact in index order.  One block per batch.
// ---------------------------------------------------------------------------
__global__ __launch_bounds__(256) void k_select(const float* __restrict__ imp,
                                                int* __restrict__ idx) {
  __shared__ float simp[S_];
  __shared__ int keep[S_];
  const int b = blockIdx.x, t = threadIdx.x;
  for (int i = t; i < S_; i += 256) simp[i] = imp[b * S_ + i];
  __syncthreads();
  for (int i = t; i < S_; i += 256) {
    const float my = simp[i];
    int rank = 0;
    for (int j = 0; j < S_; ++j) {
      const float vj = simp[j];
      rank += (vj > my) || (vj == my && j < i);
    }
    keep[i] = (rank < KEEP_) ? 1 : 0;
  }
  __syncthreads();
  for (int i = t; i < S_; i += 256) {
    if (keep[i]) {
      int pos = 0;
      for (int j = 0; j < i; ++j) pos += keep[j];
      idx[b * KEEP_ + pos] = i;
    }
  }
}

// ---------------------------------------------------------------------------
// Gather selected rows: x_sel[m] = x[b, idx[b,i]]         (block per row)
// ---------------------------------------------------------------------------
__global__ __launch_bounds__(256) void k_gather(const float* __restrict__ x,
                                                const int* __restrict__ idx,
                                                float* __restrict__ xsel) {
  const int m = blockIdx.x;
  const int b = m / KEEP_, i = m - b * KEEP_;
  const int src = idx[b * KEEP_ + i];
  const float4* s = (const float4*)(x + ((size_t)b * S_ + src) * D_);
  float4* d = (float4*)(xsel + (size_t)m * D_);
  d[threadIdx.x] = s[threadIdx.x];
}

// ---------------------------------------------------------------------------
// AdaLN parameters: p = silu(t_emb) @ W^T + b   for both LN1 and LN2.
// ---------------------------------------------------------------------------
__global__ __launch_bounds__(256) void k_adaln(
    const float* __restrict__ temb, const float* __restrict__ w1,
    const float* __restrict__ bb1, const float* __restrict__ w2,
    const float* __restrict__ bb2, float* __restrict__ p1,
    float* __restrict__ p2) {
  const int id = blockIdx.x * 256 + threadIdx.x;    // 2 * B * 2D = 16384
  const int which = id >> 13;
  const int rem = id & 8191;
  const int b = rem >> 11;
  const int n = rem & 2047;
  const float* W = which ? w2 : w1;
  const float* bb = which ? bb2 : bb1;
  const float* c = temb + b * TDIM_;
  const float* wr = W + (size_t)n * TDIM_;
  float acc = 0.f;
  for (int t = 0; t < TDIM_; ++t) {
    const float cv = c[t];
    acc += (cv / (1.f + __expf(-cv))) * wr[t];
  }
  (which ? p2 : p1)[b * 2 * D_ + n] = acc + bb[n];
}

// ---------------------------------------------------------------------------
// LayerNorm + AdaLN modulate, emit bf16:  h = LN(x)*(1+gamma) + beta
// ---------------------------------------------------------------------------
__global__ __launch_bounds__(256) void k_lnmod(const float* __restrict__ xsel,
                                               const float* __restrict__ p,
                                               __bf16* __restrict__ h) {
  __shared__ float s1[256], s2[256];
  const int m = blockIdx.x;
  const int b = m / KEEP_;
  const int t = threadIdx.x;
  const float4 v = ((const float4*)(xsel + (size_t)m * D_))[t];
  s1[t] = v.x + v.y + v.z + v.w;
  s2[t] = v.x * v.x + v.y * v.y + v.z * v.z + v.w * v.w;
  __syncthreads();
  for (int off = 128; off > 0; off >>= 1) {
    if (t < off) { s1[t] += s1[t + off]; s2[t] += s2[t + off]; }
    __syncthreads();
  }
  const float mu = s1[0] * (1.f / D_);
  const float var = s2[0] * (1.f / D_) - mu * mu;
  const float rstd = rsqrtf(var + 1e-5f);
  const float* gp = p + (size_t)b * 2 * D_;
  const float xv[4] = {v.x, v.y, v.z, v.w};
#pragma unroll
  for (int j = 0; j < 4; ++j) {
    const int n = t * 4 + j;
    h[(size_t)m * D_ + n] =
        (__bf16)(((xv[j] - mu) * rstd) * (1.f + gp[n]) + gp[D_ + n]);
  }
}

// ---------------------------------------------------------------------------
// bf16 WMMA GEMM: C[M,N] = A[M,K] @ B[N,K]^T (+bias).
// Block tile 64x128, 8 waves in a 2x4 grid; each wave computes a 32x32 tile
// (4 WMMAs per K-step of 32 against 4 b128-pair fragment loads -> 2x better
// compute/LDS ratio than a 32x16 strip).  Next K-tile is prefetched with
// global_prefetch_b8.  Fragments follow the 16-bit ISA layout: a lane's 16
// elements are two contiguous 8-element K-chunks at K=8*half and 16+8*half.
// EPI: 0 = bias -> bf16 out; 1 = exact GELU -> bf16 out;
//      2 = resid[m,n] += gate[n] * (c + bias)  (fp32 in place).
// ---------------------------------------------------------------------------
template <int EPI>
__global__ __launch_bounds__(256) void k_gemm(
    const __bf16* __restrict__ A, const __bf16* __restrict__ Bw,
    const float* __restrict__ bias, __bf16* __restrict__ obf,
    float* __restrict__ resid, const float* __restrict__ gate,
    int M, int N, int K) {
  __shared__ __align__(16) __bf16 As[64][32];
  __shared__ __align__(16) __bf16 Bs[128][32];
  const int tid = threadIdx.x;
  const int lane = tid & 31, wave = tid >> 5;
  const int half = lane >> 4, l16 = lane & 15;
  const int wm = wave >> 2, wn = wave & 3;
  const int row0 = blockIdx.y * 64, col0 = blockIdx.x * 128;
  const int lrA = tid >> 2;          // 0..63 A-tile row
  const int lcA = (tid & 3) * 8;     // K chunk of 8 bf16 (16 B)

  v8f acc[2][2];
#pragma unroll
  for (int s = 0; s < 2; ++s)
#pragma unroll
    for (int t = 0; t < 2; ++t)
#pragma unroll
      for (int k = 0; k < 8; ++k) acc[s][t][k] = 0.f;

  for (int kt = 0; kt < K; kt += 32) {
    // Stage A (64x32) and B (128x32) tiles into LDS, b128 per load.
    const int ar = row0 + lrA;
    uint4 av = make_uint4(0u, 0u, 0u, 0u);
    if (ar < M) av = *(const uint4*)(A + (size_t)ar * K + kt + lcA);
    *(uint4*)&As[lrA][lcA] = av;
#pragma unroll
    for (int j = 0; j < 2; ++j) {
      const int idx = tid + j * 256;
      const int br = idx >> 2, bc = (idx & 3) * 8;
      *(uint4*)&Bs[br][bc] =
          *(const uint4*)(Bw + (size_t)(col0 + br) * K + kt + bc);
    }
    // Prefetch next K tile while this one is consumed (global_prefetch_b8).
    if (kt + 32 < K) {
      __builtin_prefetch(A + (size_t)(ar < M ? ar : row0) * K + kt + 32 + lcA, 0, 3);
      __builtin_prefetch(Bw + (size_t)(col0 + lrA) * K + kt + 32 + lcA, 0, 3);
    }
    __syncthreads();

    Frag fa[2], fb[2];
#pragma unroll
    for (int s = 0; s < 2; ++s) {
      fa[s].u[0] = *(const uint4*)&As[wm * 32 + s * 16 + l16][8 * half];
      fa[s].u[1] = *(const uint4*)&As[wm * 32 + s * 16 + l16][16 + 8 * half];
    }
#pragma unroll
    for (int t = 0; t < 2; ++t) {
      fb[t].u[0] = *(const uint4*)&Bs[wn * 32 + t * 16 + l16][8 * half];
      fb[t].u[1] = *(const uint4*)&Bs[wn * 32 + t * 16 + l16][16 + 8 * half];
    }
#pragma unroll
    for (int s = 0; s < 2; ++s)
#pragma unroll
      for (int t = 0; t < 2; ++t)
        acc[s][t] = wmma_bf16(fa[s].v, fb[t].v, acc[s][t]);
    __syncthreads();
  }

#pragma unroll
  for (int t = 0; t < 2; ++t) {
    const int n = col0 + wn * 32 + t * 16 + l16;
    const float bv = bias ? bias[n] : 0.f;
#pragma unroll
    for (int s = 0; s < 2; ++s) {
#pragma unroll
      for (int v = 0; v < 8; ++v) {
        const int m = row0 + wm * 32 + s * 16 + v + 8 * half; // C row = v+8*half
        if (m >= M) continue;
        const float val = acc[s][t][v] + bv;
        if (EPI == 0) {
          obf[(size_t)m * N + n] = (__bf16)val;
        } else if (EPI == 1) {
          const float g = 0.5f * val * (1.f + erff(val * 0.70710678f));
          obf[(size_t)m * N + n] = (__bf16)g;
        } else {
          resid[(size_t)m * N + n] += gate[n] * val;
        }
      }
    }
  }
}

// ---------------------------------------------------------------------------
// Flash attention, one wave per (batch, head, 16-row q tile).
// Each 32-kv chunk of K and V (32x64 bf16 each) is staged into LDS with
// fully-coalesced b128 row loads (exact HBM traffic, no cacheline waste).
// scores = (Q K^T)/8 via two chained WMMAs over hd=64; online softmax with
// per-row stats reduced across the 16-lane half (C rows are M = v + 8*half);
// P is restaged through LDS (C-layout -> A-layout), then P@V via WMMA.
// ---------------------------------------------------------------------------
__global__ __launch_bounds__(32) void k_attn(const __bf16* __restrict__ qkv,
                                             __bf16* __restrict__ attnout) {
  const int qt = blockIdx.x, h = blockIdx.y, b = blockIdx.z;
  const int lane = threadIdx.x;
  const int half = lane >> 4, l16 = lane & 15;
  const size_t RS = 3 * D_;  // qkv row stride (bf16 elems)

  // Q fragments (A layout): lane row = q, two 8-elem hd chunks per fragment.
  Frag aq[2];
  const int q = qt * 16 + l16;
  const bool qv = (q < KEEP_);
  const __bf16* qb = qkv + (size_t)(b * KEEP_ + (qv ? q : 0)) * RS + h * HD_;
#pragma unroll
  for (int j = 0; j < 2; ++j) {
    if (qv) {
      aq[j].u[0] = *(const uint4*)(qb + j * 32 + 8 * half);
      aq[j].u[1] = *(const uint4*)(qb + j * 32 + 16 + 8 * half);
    } else {
      aq[j].u[0] = make_uint4(0u, 0u, 0u, 0u);
      aq[j].u[1] = make_uint4(0u, 0u, 0u, 0u);
    }
  }

  float rowmax[8], rowsum[8];
  v8f o[4];
#pragma unroll
  for (int v = 0; v < 8; ++v) { rowmax[v] = -1e30f; rowsum[v] = 0.f; }
#pragma unroll
  for (int t = 0; t < 4; ++t)
#pragma unroll
    for (int k = 0; k < 8; ++k) o[t][k] = 0.f;

  __shared__ __align__(16) __bf16 Ks[32][64];
  __shared__ __align__(16) __bf16 Vs[32][64];
  __shared__ __align__(16) __bf16 Ps[16][32];

  for (int c = 0; c < NCHUNK_; ++c) {
    const int kv0 = c * 32;

    // ---- Stage K/V chunk: lane loads one full 128B row of each ----
    __syncthreads();  // WAR vs previous iteration's fragment reads
    {
      const int kvr = kv0 + lane;
      const bool rv = (kvr < KEEP_);
      const __bf16* rp = qkv + (size_t)(b * KEEP_ + (rv ? kvr : 0)) * RS;
      const uint4 z4 = make_uint4(0u, 0u, 0u, 0u);
#pragma unroll
      for (int j = 0; j < 8; ++j) {
        uint4 kk = z4, vv = z4;
        if (rv) {
          kk = *(const uint4*)(rp + D_ + h * HD_ + j * 8);
          vv = *(const uint4*)(rp + 2 * D_ + h * HD_ + j * 8);
        }
        *(uint4*)&Ks[lane][j * 8] = kk;
        *(uint4*)&Vs[lane][j * 8] = vv;
      }
    }
    __syncthreads();

    // ---- Scores: Q @ K^T (two 16-kv subtiles) ----
    v8f sc[2];
#pragma unroll
    for (int s = 0; s < 2; ++s) {
      const int kv = kv0 + s * 16 + l16;
      const bool kvv = (kv < KEEP_);
      Frag bk0, bk1;
      bk0.u[0] = *(const uint4*)&Ks[s * 16 + l16][8 * half];
      bk0.u[1] = *(const uint4*)&Ks[s * 16 + l16][16 + 8 * half];
      bk1.u[0] = *(const uint4*)&Ks[s * 16 + l16][32 + 8 * half];
      bk1.u[1] = *(const uint4*)&Ks[s * 16 + l16][32 + 16 + 8 * half];
      v8f z;
#pragma unroll
      for (int k = 0; k < 8; ++k) z[k] = 0.f;
      z = wmma_bf16(aq[0].v, bk0.v, z);
      z = wmma_bf16(aq[1].v, bk1.v, z);
#pragma unroll
      for (int k = 0; k < 8; ++k) z[k] = kvv ? z[k] * 0.125f : -1e30f;
      sc[s] = z;
    }

    // ---- Online softmax (row stats across the 16-lane half) ----
    float t8[8];
#pragma unroll
    for (int v = 0; v < 8; ++v) t8[v] = fmaxf(sc[0][v], sc[1][v]);
    for (int off = 1; off < 16; off <<= 1)
#pragma unroll
      for (int v = 0; v < 8; ++v) t8[v] = fmaxf(t8[v], __shfl_xor(t8[v], off, 32));
    float alpha[8];
#pragma unroll
    for (int v = 0; v < 8; ++v) {
      const float mn = fmaxf(rowmax[v], t8[v]);
      alpha[v] = __expf(rowmax[v] - mn);
      rowmax[v] = mn;
    }
    float rs[8];
#pragma unroll
    for (int v = 0; v < 8; ++v) {
      const float p0 = __expf(sc[0][v] - rowmax[v]);
      const float p1 = __expf(sc[1][v] - rowmax[v]);
      sc[0][v] = p0; sc[1][v] = p1;
      rs[v] = p0 + p1;
    }
    for (int off = 1; off < 16; off <<= 1)
#pragma unroll
      for (int v = 0; v < 8; ++v) rs[v] += __shfl_xor(rs[v], off, 32);
#pragma unroll
    for (int v = 0; v < 8; ++v) rowsum[v] = rowsum[v] * alpha[v] + rs[v];
#pragma unroll
    for (int t = 0; t < 4; ++t)
#pragma unroll
      for (int k = 0; k < 8; ++k) o[t][k] *= alpha[k];

    // ---- Restage P: C-layout -> LDS -> A-layout ----
#pragma unroll
    for (int s = 0; s < 2; ++s)
#pragma unroll
      for (int v = 0; v < 8; ++v)
        Ps[v + 8 * half][s * 16 + l16] = (__bf16)sc[s][v];
    __syncthreads();
    Frag ap;
    ap.u[0] = *(const uint4*)&Ps[l16][8 * half];
    ap.u[1] = *(const uint4*)&Ps[l16][16 + 8 * half];

    // ---- O += P @ V (V B-fragments from LDS) ----
#pragma unroll
    for (int t = 0; t < 4; ++t) {
      Frag bv;
#pragma unroll
      for (int e = 0; e < 16; ++e) {
        const int kk = (e < 8) ? (8 * half + e) : (16 + 8 * half + (e - 8));
        bv.v[e] = Vs[kk][t * 16 + l16];
      }
      o[t] = wmma_bf16(ap.v, bv.v, o[t]);
    }
  }

#pragma unroll
  for (int t = 0; t < 4; ++t)
#pragma unroll
    for (int v = 0; v < 8; ++v) {
      const int qo = qt * 16 + v + 8 * half;
      if (qo < KEEP_)
        attnout[(size_t)(b * KEEP_ + qo) * D_ + h * HD_ + t * 16 + l16] =
            (__bf16)(o[t][v] / rowsum[v]);
    }
}

// ---------------------------------------------------------------------------
// Scatter processed rows back into the full output.
// ---------------------------------------------------------------------------
__global__ __launch_bounds__(256) void k_scatter(const float* __restrict__ xsel,
                                                 const int* __restrict__ idx,
                                                 float* __restrict__ out) {
  const int m = blockIdx.x;
  const int b = m / KEEP_, i = m - b * KEEP_;
  const int dst = idx[b * KEEP_ + i];
  const float4* s = (const float4*)(xsel + (size_t)m * D_);
  float4* d = (float4*)(out + ((size_t)b * S_ + dst) * D_);
  d[threadIdx.x] = s[threadIdx.x];
}

// ---------------------------------------------------------------------------
extern "C" void kernel_launch(void* const* d_in, const int* in_sizes, int n_in,
                              void* d_out, int out_size, void* d_ws,
                              size_t ws_size, hipStream_t stream) {
  const float* x        = (const float*)d_in[0];
  const float* t_emb    = (const float*)d_in[1];
  const float* tr_w1    = (const float*)d_in[2];
  const float* tr_b1    = (const float*)d_in[3];
  const float* tr_w2    = (const float*)d_in[4];
  const float* tr_b2    = (const float*)d_in[5];
  const float* ln1_w    = (const float*)d_in[6];
  const float* ln1_b    = (const float*)d_in[7];
  const float* ln2_w    = (const float*)d_in[8];
  const float* ln2_b    = (const float*)d_in[9];
  const float* in_w     = (const float*)d_in[10];
  const float* in_b     = (const float*)d_in[11];
  const float* out_w    = (const float*)d_in[12];
  const float* out_b    = (const float*)d_in[13];
  const float* ffn_w1   = (const float*)d_in[14];
  const float* ffn_b1   = (const float*)d_in[15];
  const float* ffn_w2   = (const float*)d_in[16];
  const float* ffn_b2   = (const float*)d_in[17];
  const float* gate_a   = (const float*)d_in[18];
  const float* gate_f   = (const float*)d_in[19];

  // Workspace carve-up (~155 MB total).
  char* wp = (char*)d_ws;
  auto carve = [&](size_t bytes) -> void* {
    void* r = (void*)wp;
    wp += (bytes + 255) & ~(size_t)255;
    return r;
  };
  float*  imp   = (float*) carve((size_t)B_ * S_ * 4);
  int*    idx   = (int*)   carve((size_t)B_ * KEEP_ * 4);
  float*  xsel  = (float*) carve((size_t)MTOT_ * D_ * 4);
  float*  p1    = (float*) carve((size_t)B_ * 2 * D_ * 4);
  float*  p2    = (float*) carve((size_t)B_ * 2 * D_ * 4);
  __bf16* hbuf  = (__bf16*)carve((size_t)MTOT_ * D_ * 2);
  __bf16* qkvb  = (__bf16*)carve((size_t)MTOT_ * 3 * D_ * 2);
  __bf16* attnb = (__bf16*)carve((size_t)MTOT_ * D_ * 2);
  __bf16* ffnb  = (__bf16*)carve((size_t)MTOT_ * DFF_ * 2);
  __bf16* w_in  = (__bf16*)carve((size_t)3 * D_ * D_ * 2);
  __bf16* w_out = (__bf16*)carve((size_t)D_ * D_ * 2);
  __bf16* w_f1  = (__bf16*)carve((size_t)DFF_ * D_ * 2);
  __bf16* w_f2  = (__bf16*)carve((size_t)D_ * DFF_ * 2);

  // Weight conversion to bf16 (done every call; deterministic).
  {
    int n1 = 3 * D_ * D_, n2 = D_ * D_, n3 = DFF_ * D_, n4 = D_ * DFF_;
    k_cvt<<<(n1 + 255) / 256, 256, 0, stream>>>(in_w, w_in, n1);
    k_cvt<<<(n2 + 255) / 256, 256, 0, stream>>>(out_w, w_out, n2);
    k_cvt<<<(n3 + 255) / 256, 256, 0, stream>>>(ffn_w1, w_f1, n3);
    k_cvt<<<(n4 + 255) / 256, 256, 0, stream>>>(ffn_w2, w_f2, n4);
  }

  // Token routing.
  k_importance<<<B_ * S_, 128, 0, stream>>>(x, tr_w1, tr_b1, tr_w2, tr_b2, imp);
  k_select<<<B_, 256, 0, stream>>>(imp, idx);
  k_gather<<<MTOT_, 256, 0, stream>>>(x, idx, xsel);

  // AdaLN conditioning.
  k_adaln<<<64, 256, 0, stream>>>(t_emb, ln1_w, ln1_b, ln2_w, ln2_b, p1, p2);

  const int gM = (MTOT_ + 63) / 64;  // 90

  // --- Attention path ---
  k_lnmod<<<MTOT_, 256, 0, stream>>>(xsel, p1, hbuf);
  k_gemm<0><<<dim3(3 * D_ / 128, gM), 256, 0, stream>>>(
      hbuf, w_in, in_b, qkvb, nullptr, nullptr, MTOT_, 3 * D_, D_);
  k_attn<<<dim3(QT_, HEADS_, B_), 32, 0, stream>>>(qkvb, attnb);
  k_gemm<2><<<dim3(D_ / 128, gM), 256, 0, stream>>>(
      attnb, w_out, out_b, nullptr, xsel, gate_a, MTOT_, D_, D_);

  // --- FFN path ---
  k_lnmod<<<MTOT_, 256, 0, stream>>>(xsel, p2, hbuf);
  k_gemm<1><<<dim3(DFF_ / 128, gM), 256, 0, stream>>>(
      hbuf, w_f1, ffn_b1, ffnb, nullptr, nullptr, MTOT_, DFF_, D_);
  k_gemm<2><<<dim3(D_ / 128, gM), 256, 0, stream>>>(
      ffnb, w_f2, ffn_b2, nullptr, xsel, gate_f, MTOT_, D_, DFF_);

  // --- Write output: copy x, then scatter processed rows ---
  hipMemcpyAsync(d_out, (const void*)x, (size_t)B_ * S_ * D_ * sizeof(float),
                 hipMemcpyDeviceToDevice, stream);
  k_scatter<<<MTOT_, 256, 0, stream>>>(xsel, idx, (float*)d_out);
}
